// Part_Graph_41927470743597
// MI455X (gfx1250) — matrix operand order repaired and checked by reading, not compiled, not run on hardware
//
#include <hip/hip_runtime.h>

#define NB    4
#define CIN   256
#define LPIX  4096
#define NTOT  16384          // = 2^14
#define NHID  10
#define KPAD  288
#define FEATC 264
#define EPSB  1e-5f

typedef __attribute__((ext_vector_type(16))) _Float16 v16h;
typedef __attribute__((ext_vector_type(8)))  _Float16 v8h;
typedef __attribute__((ext_vector_type(8)))  float    v8f;

__device__ __forceinline__ float sigm(float x){ return 1.0f/(1.0f+expf(-x)); }

__device__ __forceinline__ float coord_feat(int c, int l){
  int h = l >> 6, w = l & 63;
  const float inv = 1.0f/64.0f;
  float xmin = (float)w*(2.0f*inv) - 1.0f;
  float xmax = (float)(w+1)*(2.0f*inv) - 1.0f;
  float ymin = (float)h*(2.0f*inv) - 1.0f;
  float ymax = (float)(h+1)*(2.0f*inv) - 1.0f;
  switch(c){
    case 0: return xmin;
    case 1: return ymin;
    case 2: return xmax;
    case 3: return ymax;
    case 4: return 0.5f*(xmin+xmax);
    case 5: return 0.5f*(ymin+ymax);
    default: return inv;   // 6,7: 1/W, 1/H
  }
}

// ---------------- prep: f16 feature matrix PACKED in WMMA B-fragment order ----------------
// Layout: Xf[((kb*NTOT + col)*32) + kk], kb = k/32, kk = k%32.
// A lane of the GEMM reads halves kk = hi*16 .. hi*16+15 for its column: one
// contiguous 32-byte run (2 x global_load_b128), wave covers 1KB contiguous.
__global__ void k_prep_xf(const float* __restrict__ xp, _Float16* __restrict__ Xf){
  int idx = blockIdx.x*blockDim.x + threadIdx.x;
  if (idx >= KPAD*NTOT) return;
  int kk   = idx & 31;
  int rest = idx >> 5;            // kb*NTOT + col
  int col  = rest & (NTOT-1);
  int kb   = rest >> 14;
  int k    = kb*32 + kk;
  int n = col >> 12, l = col & 4095;
  float v = 0.f;
  if (k < CIN)        v = xp[((size_t)n*CIN + k)*LPIX + l];
  else if (k < FEATC) v = coord_feat(k-CIN, l);
  Xf[idx] = (_Float16)v;
}

__global__ void k_prep_w(const float* __restrict__ W_key, const float* __restrict__ W_att,
                         const float* __restrict__ W_proj,
                         _Float16* __restrict__ Wk16, _Float16* __restrict__ Wc16){
  int tid = blockIdx.x*blockDim.x + threadIdx.x;
  int stride = gridDim.x*blockDim.x;
  for (int i = tid; i < 64*KPAD; i += stride){
    int o = i / KPAD, c = i - o*KPAD;
    Wk16[i] = (_Float16)((c < FEATC) ? W_key[o*FEATC + c] : 0.f);
  }
  // combined GEMM weights: rows 0..11 = W_att[:,1:257], rows 16..25 = W_proj
  for (int i = tid; i < 32*CIN; i += stride){
    int r = i / CIN, c = i - r*CIN;
    float v = 0.f;
    if (r < 12)                 v = W_att[r*257 + 1 + c];
    else if (r >= 16 && r < 26) v = W_proj[(r-16)*CIN + c];
    Wc16[i] = (_Float16)v;
  }
}

// ---------------- softmax over L of p_att -> norm [24][LPIX] ----------------
__global__ void k_softmax_norm(const float* __restrict__ p_atts, float* __restrict__ norm){
  __shared__ float red[256];
  int pn = blockIdx.x;                       // p*4+n, p in 0..5
  int p = pn >> 2, n = pn & 3;
  const float* src = p_atts + ((size_t)(p+1)*NB + n)*LPIX;
  int tid = threadIdx.x;
  float mx = -1e30f;
  for (int l = tid; l < LPIX; l += 256) mx = fmaxf(mx, src[l]);
  red[tid] = mx; __syncthreads();
  for (int s = 128; s > 0; s >>= 1){ if (tid < s) red[tid] = fmaxf(red[tid], red[tid+s]); __syncthreads(); }
  mx = red[0]; __syncthreads();
  float sm = 0.f;
  for (int l = tid; l < LPIX; l += 256) sm += expf(src[l]-mx);
  red[tid] = sm; __syncthreads();
  for (int s = 128; s > 0; s >>= 1){ if (tid < s) red[tid] += red[tid+s]; __syncthreads(); }
  float inv = 1.0f/red[0];
  float* dst = norm + (size_t)pn*LPIX;
  for (int l = tid; l < LPIX; l += 256) dst[l] = expf(src[l]-mx)*inv;
}

// ---------------- centroid (264ch weighted sum) + q = W_q*cen + b_q ----------------
__global__ void k_centroid_q(const float* __restrict__ xp, const float* __restrict__ norm,
                             const float* __restrict__ W_q, const float* __restrict__ b_q,
                             float* __restrict__ qout){
  __shared__ float nsh[LPIX];
  __shared__ float cen[FEATC];
  int pn = blockIdx.x; int n = pn & 3;
  int tid = threadIdx.x;                     // blockDim = 288
  for (int l = tid; l < LPIX; l += blockDim.x) nsh[l] = norm[(size_t)pn*LPIX + l];
  __syncthreads();
  if (tid < FEATC){
    float s = 0.f;
    if (tid < CIN){
      const float* row = xp + ((size_t)n*CIN + tid)*LPIX;
      for (int l = 0; l < LPIX; ++l) s += row[l]*nsh[l];
    } else {
      int c = tid - CIN;
      for (int l = 0; l < LPIX; ++l) s += coord_feat(c, l)*nsh[l];
    }
    cen[tid] = s;
  }
  __syncthreads();
  if (tid < 64){
    float s = b_q[tid];
    const float* wr = W_q + tid*FEATC;
    for (int f = 0; f < FEATC; ++f) s += wr[f]*cen[f];
    qout[(size_t)pn*64 + tid] = s;
  }
}

// ---------------- WMMA GEMM: C[M][NTOT] = A(f16 MxK rowmajor) * Bpacked ----------------
// One wave computes a 16x64 strip: A fragment reused across 4 WMMAs per k-step.
__global__ void k_wmma_gemm(const _Float16* __restrict__ A, const _Float16* __restrict__ Bp,
                            float* __restrict__ C, int K){
  int lane = threadIdx.x & 31;
  int hi   = lane >> 4;            // half-wave select
  int m16  = lane & 15;
  int ntile = blockIdx.x * 64;
  int mtile = blockIdx.y * 16;
  v8f acc0 = {}, acc1 = {}, acc2 = {}, acc3 = {};
  for (int k0 = 0; k0 < K; k0 += 32){
    // A 16x32 f16 fragment: lanes 0-15 K=[k0..k0+15], lanes 16-31 K=[k0+16..k0+31]
    const _Float16* arow = A + (size_t)(mtile + m16)*K + k0 + hi*16;
    v8h alo = *(const v8h*)(arow);
    v8h ahi = *(const v8h*)(arow + 8);
    v16h a = __builtin_shufflevector(alo, ahi, 0,1,2,3,4,5,6,7,8,9,10,11,12,13,14,15);
    // B fragments from packed layout: 32 contiguous bytes per lane per tile
    const _Float16* bbase = Bp + ((size_t)(k0 >> 5)*NTOT + ntile + m16)*32 + hi*16;
    v16h b0 = *(const v16h*)(bbase);
    v16h b1 = *(const v16h*)(bbase + 16*32);
    v16h b2 = *(const v16h*)(bbase + 32*32);
    v16h b3 = *(const v16h*)(bbase + 48*32);
    __builtin_prefetch(bbase + (size_t)NTOT*32, 0, 1);
    acc0 = __builtin_amdgcn_wmma_f32_16x16x32_f16(false, a, false, b0, (short)0, acc0, false, false);
    acc1 = __builtin_amdgcn_wmma_f32_16x16x32_f16(false, a, false, b1, (short)0, acc1, false, false);
    acc2 = __builtin_amdgcn_wmma_f32_16x16x32_f16(false, a, false, b2, (short)0, acc2, false, false);
    acc3 = __builtin_amdgcn_wmma_f32_16x16x32_f16(false, a, false, b3, (short)0, acc3, false, false);
  }
  // C 16x16 f32 layout: VGPR r -> row mtile + hi*8 + r, col tile_base + (lane&15)
  float* crow = C + (size_t)(mtile + hi*8)*NTOT + ntile + m16;
  #pragma unroll
  for (int r = 0; r < 8; ++r){
    crow[(size_t)r*NTOT +  0] = acc0[r];
    crow[(size_t)r*NTOT + 16] = acc1[r];
    crow[(size_t)r*NTOT + 32] = acc2[r];
    crow[(size_t)r*NTOT + 48] = acc3[r];
  }
}

// ---------------- energy -> attn -> s0 ; logits (+bias+s0 channel) -> out + softmax ----------------
__global__ void k_attn_logits(const float* __restrict__ qv, const float* __restrict__ keyf,
                              const float* __restrict__ b_key, const float* __restrict__ p_atts,
                              const float* __restrict__ zy, const float* __restrict__ W_att,
                              const float* __restrict__ b_att,
                              float* __restrict__ s0, float* __restrict__ attsoft,
                              float* __restrict__ out_att){
  __shared__ float qsh[6*NB*64];
  __shared__ float bk[64];
  int tid = threadIdx.x;
  for (int i = tid; i < 6*NB*64; i += 256) qsh[i] = qv[i];
  for (int i = tid; i < 64; i += 256) bk[i] = b_key[i];
  __syncthreads();
  int col = blockIdx.x*256 + tid;
  int n = col >> 12, l = col & 4095;
  float e[6];
  #pragma unroll
  for (int p = 0; p < 6; ++p) e[p] = 0.f;
  for (int o = 0; o < 64; ++o){
    float kv = keyf[(size_t)o*NTOT + col] + bk[o];
    #pragma unroll
    for (int p = 0; p < 6; ++p) e[p] += qsh[(p*NB+n)*64+o]*kv;
  }
  float s0v[6];
  #pragma unroll
  for (int p = 0; p < 6; ++p){
    float attn = sigm(e[p]);
    float pa = p_atts[((size_t)(p+1)*NB + n)*LPIX + l];
    s0v[p] = attn*(1.f - pa);                 // == F_dep_att[p]
    s0[(size_t)p*NTOT + col] = s0v[p];
  }
  const int OFFP[6] = {0,1,6,7,8,10};
  const int DEGP[6] = {1,5,1,1,2,2};
  for (int p = 0; p < 6; ++p){
    int d = DEGP[p], off = OFFP[p];
    float lg[5];
    float mx = -1e30f;
    for (int j = 0; j < d; ++j){
      int row = off + j;
      float v = zy[(size_t)row*NTOT + col] + W_att[row*257]*s0v[p] + b_att[row];
      lg[j] = v;
      out_att[((size_t)n*12 + row)*LPIX + l] = v;
      mx = fmaxf(mx, v);
    }
    float sm = 0.f;
    for (int j = 0; j < d; ++j){ lg[j] = expf(lg[j]-mx); sm += lg[j]; }
    float inv = 1.f/sm;
    for (int j = 0; j < d; ++j) attsoft[(size_t)(off+j)*NTOT + col] = lg[j]*inv;
  }
}

// deterministic per-block reduction of (v1,v2) -> partial arrays
__device__ __forceinline__ void blockReduce2(float v1, float v2, float* red,
                                             float* o1, float* o2){
  int tid = threadIdx.x;
  red[tid] = v1; red[256+tid] = v2; __syncthreads();
  for (int s = 128; s > 0; s >>= 1){
    if (tid < s){ red[tid] += red[tid+s]; red[256+tid] += red[256+tid+s]; }
    __syncthreads();
  }
  if (tid == 0){ *o1 = red[0]; *o2 = red[256]; }
  __syncthreads();
}

// ---------------- edge stage 1: BN stats of msg_pre = scal * Y ----------------
__global__ void k_edge_stats1(const float* __restrict__ s0, const float* __restrict__ attsoft,
                              const float* __restrict__ zy, int u, int attRow,
                              float* __restrict__ st){     // st: [64][10] sums, [64][10] sumsq
  __shared__ float red[512];
  int tid = threadIdx.x;
  int col = blockIdx.x*256 + tid;
  float scal = s0[(size_t)u*NTOT + col]*attsoft[(size_t)attRow*NTOT + col];
  for (int o = 0; o < NHID; ++o){
    float v = scal*zy[(size_t)(16+o)*NTOT + col];
    blockReduce2(v, v*v, red, &st[blockIdx.x*NHID + o], &st[64*NHID + blockIdx.x*NHID + o]);
  }
}

// ---------------- edge stage 2: normalize + relu -> msg[n][10][hw] ----------------
__global__ void k_edge_msg(const float* __restrict__ s0, const float* __restrict__ attsoft,
                           const float* __restrict__ zy, const float* __restrict__ st,
                           const float* __restrict__ g_proj, const float* __restrict__ bt_proj,
                           int u, int attRow, float* __restrict__ msg){
  __shared__ float mu[NHID], rs[NHID];
  int tid = threadIdx.x;
  if (tid < NHID){
    float s = 0.f, q = 0.f;
    for (int b = 0; b < 64; ++b){ s += st[b*NHID + tid]; q += st[64*NHID + b*NHID + tid]; }
    float m = s*(1.0f/NTOT); mu[tid] = m; rs[tid] = rsqrtf(q*(1.0f/NTOT) - m*m + EPSB);
  }
  __syncthreads();
  int col = blockIdx.x*256 + tid;
  int n = col >> 12, l = col & 4095;
  float scal = s0[(size_t)u*NTOT + col]*attsoft[(size_t)attRow*NTOT + col];
  for (int o = 0; o < NHID; ++o){
    float v = scal*zy[(size_t)(16+o)*NTOT + col];
    float y = (v - mu[o])*rs[o]*g_proj[o] + bt_proj[o];
    msg[((size_t)n*NHID + o)*LPIX + l] = fmaxf(y, 0.f);
  }
}

// ---------------- edge stage 3: conv3x3 over [msg ; p_nodes[v]] + BN2 stats ----------------
__global__ void k_edge_conv3(const float* __restrict__ msg, const float* __restrict__ p_nodes,
                             const float* __restrict__ W_rel1, int v,
                             float* __restrict__ y20, float* __restrict__ st2){
  __shared__ float wsh[20*20*9];
  __shared__ float red[512];
  int tid = threadIdx.x;
  for (int i = tid; i < 3600; i += 256) wsh[i] = W_rel1[i];
  __syncthreads();
  int pix = blockIdx.x*256 + tid;
  int n = pix >> 12, hw = pix & 4095;
  int h = hw >> 6, w = hw & 63;
  float acc[20];
  #pragma unroll
  for (int co = 0; co < 20; ++co) acc[co] = 0.f;
  for (int ci = 0; ci < 20; ++ci){
    const float* src = (ci < NHID)
      ? (msg + ((size_t)n*NHID + ci)*LPIX)
      : (p_nodes + (((size_t)v*NB + n)*NHID + (ci-NHID))*LPIX);
    for (int ky = 0; ky < 3; ++ky){
      int h2 = h + ky - 1;
      if (h2 < 0 || h2 > 63) continue;
      for (int kx = 0; kx < 3; ++kx){
        int w2 = w + kx - 1;
        if (w2 < 0 || w2 > 63) continue;
        float xv = src[h2*64 + w2];
        int wb = ci*9 + ky*3 + kx;
        #pragma unroll
        for (int co = 0; co < 20; ++co) acc[co] += wsh[co*180 + wb]*xv;
      }
    }
  }
  for (int co = 0; co < 20; ++co){
    y20[((size_t)n*20 + co)*LPIX + hw] = acc[co];
    blockReduce2(acc[co], acc[co]*acc[co], red,
                 &st2[blockIdx.x*20 + co], &st2[64*20 + blockIdx.x*20 + co]);
  }
}

// ---------------- edge stage 4: BN2+relu -> conv1x1 rel2 -> sigmoid -> += msgs[v] ----------------
__global__ void k_edge_finish(const float* __restrict__ y20, const float* __restrict__ st2,
                              const float* __restrict__ g_rel1, const float* __restrict__ bt_rel1,
                              const float* __restrict__ W_rel2, const float* __restrict__ b_rel2,
                              int v, float* __restrict__ msgs){
  __shared__ float wsh[200];
  __shared__ float mu[20], rs[20];
  int tid = threadIdx.x;
  for (int i = tid; i < 200; i += 256) wsh[i] = W_rel2[i];
  if (tid < 20){
    float s = 0.f, q = 0.f;
    for (int b = 0; b < 64; ++b){ s += st2[b*20 + tid]; q += st2[64*20 + b*20 + tid]; }
    float m = s*(1.0f/NTOT); mu[tid] = m; rs[tid] = rsqrtf(q*(1.0f/NTOT) - m*m + EPSB);
  }
  __syncthreads();
  int pix = blockIdx.x*256 + tid;
  int n = pix >> 12, hw = pix & 4095;
  float xr[20];
  for (int c = 0; c < 20; ++c){
    float val = y20[((size_t)n*20 + c)*LPIX + hw];
    float yv = (val - mu[c])*rs[c]*g_rel1[c] + bt_rel1[c];
    xr[c] = fmaxf(yv, 0.f);
  }
  for (int o = 0; o < NHID; ++o){
    float s = b_rel2[o];
    for (int c = 0; c < 20; ++c) s += wsh[o*20 + c]*xr[c];
    msgs[(((size_t)v*NB + n)*NHID + o)*LPIX + hw] += sigm(s);
  }
}

// ---------------- GRU stage 1: gates, candidate pre-BN + stats ----------------
__global__ void k_gru1(const float* __restrict__ msgs, const float* __restrict__ p_nodes,
                       const float* __restrict__ Wg, const float* __restrict__ bg,
                       const float* __restrict__ Wc, int i,
                       float* __restrict__ cnm, float* __restrict__ zbuf,
                       float* __restrict__ st3){
  __shared__ float red[512];
  int tid = threadIdx.x;
  int pix = blockIdx.x*256 + tid;
  int n = pix >> 12, hw = pix & 4095;
  float m[NHID], hc[NHID];
  for (int c = 0; c < NHID; ++c){
    m[c]  = msgs[(((size_t)i*NB + n)*NHID + c)*LPIX + hw];
    hc[c] = p_nodes[(((size_t)(i+1)*NB + n)*NHID + c)*LPIX + hw];
  }
  float g0 = bg[i*2+0], g1 = bg[i*2+1];
  for (int c = 0; c < NHID; ++c){
    g0 += Wg[(i*2+0)*20 + c]*m[c] + Wg[(i*2+0)*20 + 10 + c]*hc[c];
    g1 += Wg[(i*2+1)*20 + c]*m[c] + Wg[(i*2+1)*20 + 10 + c]*hc[c];
  }
  float r = sigm(g0), z = sigm(g1);
  zbuf[pix] = z;
  for (int o = 0; o < NHID; ++o){
    float s = 0.f;
    const float* wr = Wc + ((size_t)i*NHID + o)*20;
    for (int c = 0; c < NHID; ++c) s += wr[c]*m[c] + wr[10+c]*(r*hc[c]);
    cnm[((size_t)n*NHID + o)*LPIX + hw] = s;
    blockReduce2(s, s*s, red, &st3[blockIdx.x*NHID + o], &st3[64*NHID + blockIdx.x*NHID + o]);
  }
}

// ---------------- GRU stage 2: BN + leaky_relu + blend -> out_nodes[i+1] ----------------
__global__ void k_gru2(const float* __restrict__ cnm, const float* __restrict__ zbuf,
                       const float* __restrict__ p_nodes, const float* __restrict__ st3,
                       const float* __restrict__ g_can, const float* __restrict__ bt_can,
                       int i, float* __restrict__ out_nodes){
  __shared__ float mu[NHID], rs[NHID];
  int tid = threadIdx.x;
  if (tid < NHID){
    float s = 0.f, q = 0.f;
    for (int b = 0; b < 64; ++b){ s += st3[b*NHID + tid]; q += st3[64*NHID + b*NHID + tid]; }
    float mm = s*(1.0f/NTOT); mu[tid] = mm; rs[tid] = rsqrtf(q*(1.0f/NTOT) - mm*mm + EPSB);
  }
  __syncthreads();
  int pix = blockIdx.x*256 + tid;
  int n = pix >> 12, hw = pix & 4095;
  float z = zbuf[pix];
  for (int o = 0; o < NHID; ++o){
    float s = cnm[((size_t)n*NHID + o)*LPIX + hw];
    float yv = (s - mu[o])*rs[o]*g_can[i*NHID + o] + bt_can[i*NHID + o];
    yv = (yv > 0.f) ? yv : 0.01f*yv;
    float hc = p_nodes[(((size_t)(i+1)*NB + n)*NHID + o)*LPIX + hw];
    out_nodes[(((size_t)(i+1)*NB + n)*NHID + o)*LPIX + hw] = (1.f - z)*hc + z*yv;
  }
}

__global__ void k_copy(const float* __restrict__ src, float* __restrict__ dst, int nelem){
  int i = blockIdx.x*blockDim.x + threadIdx.x;
  if (i < nelem) dst[i] = src[i];
}

extern "C" void kernel_launch(void* const* d_in, const int* in_sizes, int n_in,
                              void* d_out, int out_size, void* d_ws, size_t ws_size,
                              hipStream_t stream){
  (void)in_sizes; (void)n_in; (void)out_size; (void)ws_size;
  const float* xp      = (const float*)d_in[0];
  const float* p_nodes = (const float*)d_in[1];
  const float* p_atts  = (const float*)d_in[2];
  const float* W_key   = (const float*)d_in[3];
  const float* b_key   = (const float*)d_in[4];
  const float* W_q     = (const float*)d_in[5];
  const float* b_q     = (const float*)d_in[6];
  const float* W_att   = (const float*)d_in[7];
  const float* b_att   = (const float*)d_in[8];
  const float* W_proj  = (const float*)d_in[9];
  const float* g_proj  = (const float*)d_in[10];
  const float* bt_proj = (const float*)d_in[11];
  const float* W_rel1  = (const float*)d_in[12];
  const float* g_rel1  = (const float*)d_in[13];
  const float* bt_rel1 = (const float*)d_in[14];
  const float* W_rel2  = (const float*)d_in[15];
  const float* b_rel2  = (const float*)d_in[16];
  const float* Wg      = (const float*)d_in[17];
  const float* bg      = (const float*)d_in[18];
  const float* Wc      = (const float*)d_in[19];
  const float* g_can   = (const float*)d_in[20];
  const float* bt_can  = (const float*)d_in[21];

  float* out_nodes = (float*)d_out;                       // (7,4,10,64,64)
  float* out_att   = out_nodes + (size_t)7*NB*NHID*LPIX;  // (4,12,64,64)

  char* base = (char*)d_ws;
  size_t off = 0;
  auto alloc = [&](size_t bytes)->char*{
    char* p = base + off; off += (bytes + 255) & ~(size_t)255; return p;
  };
  float*    s_norm = (float*)alloc((size_t)24*LPIX*4);
  float*    s_q    = (float*)alloc((size_t)24*64*4);
  float*    s_s0   = (float*)alloc((size_t)6*NTOT*4);
  float*    s_att  = (float*)alloc((size_t)12*NTOT*4);
  float*    s_keyf = (float*)alloc((size_t)64*NTOT*4);
  float*    s_zy   = (float*)alloc((size_t)32*NTOT*4);
  _Float16* s_xf   = (_Float16*)alloc((size_t)KPAD*NTOT*2);
  _Float16* s_wk   = (_Float16*)alloc((size_t)64*KPAD*2);
  _Float16* s_wc   = (_Float16*)alloc((size_t)32*CIN*2);
  float*    s_msgs = (float*)alloc((size_t)6*NB*NHID*LPIX*4);
  float*    s_msg  = (float*)alloc((size_t)NB*NHID*LPIX*4);
  float*    s_y20  = (float*)alloc((size_t)NB*20*LPIX*4);
  float*    s_cnm  = (float*)alloc((size_t)NB*NHID*LPIX*4);
  float*    s_z    = (float*)alloc((size_t)NTOT*4);
  float*    s_st1  = (float*)alloc((size_t)2*64*NHID*4);
  float*    s_st2  = (float*)alloc((size_t)2*64*20*4);
  float*    s_st3  = (float*)alloc((size_t)2*64*NHID*4);

  // feature matrix (packed B-fragment order) + f16 weights, softmax-norm, centroids, q
  k_prep_xf<<<(KPAD*NTOT + 255)/256, 256, 0, stream>>>(xp, s_xf);
  k_prep_w<<<64, 256, 0, stream>>>(W_key, W_att, W_proj, s_wk, s_wc);
  k_softmax_norm<<<24, 256, 0, stream>>>(p_atts, s_norm);
  k_centroid_q<<<24, 288, 0, stream>>>(xp, s_norm, W_q, b_q, s_q);

  // WMMA GEMMs: key_f (64 x 16384, K=288) and fused [logits ; proj] (32 x 16384, K=256)
  dim3 gk(NTOT/64, 64/16);
  k_wmma_gemm<<<gk, 32, 0, stream>>>(s_wk, s_xf, s_keyf, KPAD);
  dim3 gz(NTOT/64, 32/16);
  k_wmma_gemm<<<gz, 32, 0, stream>>>(s_wc, s_xf, s_zy, CIN);

  // attention maps, s0 = attn*(1-p_att), logits (-> output) + per-part softmax
  k_attn_logits<<<NTOT/256, 256, 0, stream>>>(s_q, s_keyf, b_key, p_atts, s_zy,
                                              W_att, b_att, s_s0, s_att, out_att);

  hipMemsetAsync(s_msgs, 0, (size_t)6*NB*NHID*LPIX*4, stream);

  // edges (u, v, att-row = OFF[u]+ch)
  const int EU[12] = {0,1,1,1,1,1,2,3,4,4,5,5};
  const int EV[12] = {1,0,2,3,4,5,1,1,1,5,1,4};
  const int ER[12] = {0,1,2,3,4,5,6,7,8,9,10,11};
  for (int e = 0; e < 12; ++e){
    k_edge_stats1<<<NTOT/256, 256, 0, stream>>>(s_s0, s_att, s_zy, EU[e], ER[e], s_st1);
    k_edge_msg  <<<NTOT/256, 256, 0, stream>>>(s_s0, s_att, s_zy, s_st1,
                                               g_proj, bt_proj, EU[e], ER[e], s_msg);
    k_edge_conv3<<<NTOT/256, 256, 0, stream>>>(s_msg, p_nodes, W_rel1, EV[e], s_y20, s_st2);
    k_edge_finish<<<NTOT/256, 256, 0, stream>>>(s_y20, s_st2, g_rel1, bt_rel1,
                                                W_rel2, b_rel2, EV[e], s_msgs);
  }

  // xp_new[0] = p_nodes[0]; GRU update for parts 1..6
  k_copy<<<(NB*NHID*LPIX + 255)/256, 256, 0, stream>>>(p_nodes, out_nodes, NB*NHID*LPIX);
  for (int i = 0; i < 6; ++i){
    k_gru1<<<NTOT/256, 256, 0, stream>>>(s_msgs, p_nodes, Wg, bg, Wc, i, s_cnm, s_z, s_st3);
    k_gru2<<<NTOT/256, 256, 0, stream>>>(s_cnm, s_z, p_nodes, s_st3, g_can, bt_can,
                                         i, out_nodes);
  }
}
